// VI_domain_44942537785465
// MI455X (gfx1250) — matrix operand order repaired and checked by reading, not compiled, not run on hardware
//
#include <hip/hip_runtime.h>
#include <hip/hip_bf16.h>
#include <math.h>

typedef __attribute__((ext_vector_type(16))) _Float16 v16h;
typedef __attribute__((ext_vector_type(8)))  float    v8f;

#define DOCN 8192
#define VOCN 512
#define TN   16
#define DN   10
#define PN   16
#define SN   512

// ---- workspace layout (float units); total ~107,600 floats (~430 KB) ----
#define OFF_DP     0                      // [8192*10] domain_prob
#define OFF_Q      (OFF_DP + DOCN*DN)     // [8192]    q[s] = sum_d dp^2
#define OFF_RMAX   (OFF_Q + DOCN)         // [8192]    kernel row max
#define OFF_RINVZ  (OFF_RMAX + DOCN)      // [8192]    kernel row 1/sum(exp)
#define OFF_ENT    (OFF_RINVZ + DOCN)     // [32]      entropy block partials
#define OFF_G1     (OFF_ENT + 32)         // [1024]    g1 block partials (sum_d dp*sq)
#define OFF_GRAPH  (OFF_G1 + 1024)        // [48]      per-partition {tr_deg, tr_adj, f2}

// -------------------------------------------------------------------------
// 1) per-doc softmax over D=10: dp, q, entropy partials
// -------------------------------------------------------------------------
__global__ __launch_bounds__(256)
void domain_kernel(const float* __restrict__ post_domain, float* __restrict__ ws) {
  int doc = blockIdx.x * blockDim.x + threadIdx.x;
  const float* pd = post_domain + (size_t)doc * DN;
  float x[DN];
  float m = -__builtin_inff();
#pragma unroll
  for (int d = 0; d < DN; ++d) { x[d] = pd[d]; m = fmaxf(m, x[d]); }
  float Z = 0.f;
#pragma unroll
  for (int d = 0; d < DN; ++d) Z += __expf(x[d] - m);
  float logZ = __logf(Z);
  float ent = 0.f, q = 0.f;
  float* dp = ws + OFF_DP + (size_t)doc * DN;
#pragma unroll
  for (int d = 0; d < DN; ++d) {
    float lp = x[d] - m - logZ;
    float p  = __expf(lp);
    dp[d] = p;
    q   += p * p;
    ent -= p * lp;
  }
  ws[OFF_Q + doc] = q;

  __shared__ float red[256];
  red[threadIdx.x] = ent;
  __syncthreads();
  for (int s = 128; s > 0; s >>= 1) {
    if (threadIdx.x < s) red[threadIdx.x] += red[threadIdx.x + s];
    __syncthreads();
  }
  if (threadIdx.x == 0) ws[OFF_ENT + blockIdx.x] = red[0];
}

// -------------------------------------------------------------------------
// 2) per-kernel-row softmax stats (max, 1/sumexp). wave per row.
// -------------------------------------------------------------------------
__global__ __launch_bounds__(256)
void rowstats_kernel(const float* __restrict__ kernels, float* __restrict__ ws) {
  int wave = threadIdx.x >> 5, lane = threadIdx.x & 31;
  int row  = blockIdx.x * 8 + wave;                  // 0..8191 (= p*512 + i)
  const float* kr = kernels + (size_t)row * SN;
  float v[16];
  float m = -__builtin_inff();
#pragma unroll
  for (int k = 0; k < 16; ++k) { v[k] = kr[lane + 32 * k]; m = fmaxf(m, v[k]); }
#pragma unroll
  for (int off = 16; off; off >>= 1) m = fmaxf(m, __shfl_xor(m, off, 32));
  float Z = 0.f;
#pragma unroll
  for (int k = 0; k < 16; ++k) Z += __expf(v[k] - m);
#pragma unroll
  for (int off = 16; off; off >>= 1) Z += __shfl_xor(Z, off, 32);
  if (lane == 0) {
    ws[OFF_RMAX + row]  = m;
    ws[OFF_RINVZ + row] = 1.f / Z;
  }
}

// -------------------------------------------------------------------------
// 3) deconvolution + Gaussian term: wave per doc, streams post_topic once.
// -------------------------------------------------------------------------
__global__ __launch_bounds__(256)
void decon_kernel(const float* __restrict__ post_topic,
                  const float* __restrict__ sp_count,
                  const float* __restrict__ exp_log_dtot,
                  float* __restrict__ ws) {
  int wave = threadIdx.x >> 5, lane = threadIdx.x & 31;
  int doc  = blockIdx.x * 8 + wave;
  float acc[TN];
#pragma unroll
  for (int t = 0; t < TN; ++t) acc[t] = 0.f;

  const float* base = post_topic + (size_t)doc * VOCN * TN;
  const float* cnt  = sp_count  + (size_t)doc * VOCN;
#pragma unroll 4
  for (int k = 0; k < 16; ++k) {
    int voc = lane + 32 * k;
    const float4* p4 = (const float4*)(base + (size_t)voc * TN);
    float4 a = p4[0], b = p4[1], c = p4[2], d = p4[3];
    float x[16] = {a.x,a.y,a.z,a.w, b.x,b.y,b.z,b.w,
                   c.x,c.y,c.z,c.w, d.x,d.y,d.z,d.w};
    float m = x[0];
#pragma unroll
    for (int t = 1; t < 16; ++t) m = fmaxf(m, x[t]);
    float e[16], Z = 0.f;
#pragma unroll
    for (int t = 0; t < 16; ++t) { e[t] = __expf(x[t] - m); Z += e[t]; }
    float sc = cnt[voc] / Z;
#pragma unroll
    for (int t = 0; t < 16; ++t) acc[t] += e[t] * sc;
  }
#pragma unroll
  for (int t = 0; t < 16; ++t)
#pragma unroll
    for (int off = 16; off; off >>= 1) acc[t] += __shfl_xor(acc[t], off, 32);

  float tot = 0.f;
#pragma unroll
  for (int t = 0; t < 16; ++t) tot += acc[t];
  float inv = 1.f / (tot + 1e-20f);

  float g = 0.f;
  if (lane < DN) {
    const float* E = exp_log_dtot + lane * TN;
    float sq = 0.f;
#pragma unroll
    for (int t = 0; t < 16; ++t) { float df = acc[t] * inv - E[t]; sq += df * df; }
    g = ws[OFF_DP + (size_t)doc * DN + lane] * sq;
  }
#pragma unroll
  for (int off = 16; off; off >>= 1) g += __shfl_xor(g, off, 32);

  __shared__ float red[8];
  if (lane == 0) red[wave] = g;
  __syncthreads();
  if (threadIdx.x == 0) {
    float s = 0.f;
    for (int w = 0; w < 8; ++w) s += red[w];
    ws[OFF_G1 + blockIdx.x] = s;   // g1 = -mean(this); sign handled in final
  }
}

// -------------------------------------------------------------------------
// 4) graph Laplacian + f2 via WMMA. One workgroup per partition.
//    accA = A(adjacency) @ [c | q | 0..] ; accP = softmax(K) @ [pp | 0..]
// -------------------------------------------------------------------------
__global__ __launch_bounds__(256)
void graph_kernel(const float* __restrict__ kernels,
                  const float* __restrict__ prior_pi,
                  float* __restrict__ ws) {
  int p = blockIdx.x;
  int tid = threadIdx.x;
  int wave = tid >> 5, lane = tid & 31;
  int h = lane >> 4, n = lane & 15;

  __shared__ _Float16 lds_c[SN * 16];   // cols 0..9 = dp, col 10 = q, rest 0
  __shared__ _Float16 lds_pp[SN * 16];  // cols 0..9 = prior_pi, rest 0
  __shared__ float red[8][3];

  for (int idx = tid; idx < SN * 16; idx += 256) {
    int j = idx >> 4, col = idx & 15;
    int doc = p * SN + j;
    float cv = 0.f, pv = 0.f;
    if (col < DN) {
      cv = ws[OFF_DP + (size_t)doc * DN + col];
      pv = prior_pi[(size_t)doc * DN + col];
    } else if (col == DN) {
      cv = ws[OFF_Q + doc];               // q rides in column 10
    }
    lds_c[idx]  = (_Float16)cv;
    lds_pp[idx] = (_Float16)pv;
  }
  __syncthreads();

  v8f accA[4], accP[4];
#pragma unroll
  for (int t = 0; t < 4; ++t) {
    accA[t] = (v8f){0.f,0.f,0.f,0.f,0.f,0.f,0.f,0.f};
    accP[t] = (v8f){0.f,0.f,0.f,0.f,0.f,0.f,0.f,0.f};
  }

  float rm[4], rz[4];
#pragma unroll
  for (int t = 0; t < 4; ++t) {
    int m0 = (wave + 8 * t) * 16;
    int grow = p * SN + m0 + n;           // A-frag row for this lane = n
    rm[t] = ws[OFF_RMAX + grow];
    rz[t] = ws[OFF_RINVZ + grow];
  }

  for (int kc = 0; kc < 16; ++kc) {
    int k0 = kc * 32;
    // B fragments: element e -> K = (e>>3)*16 + h*8 + (e&7), N = n
    v16h bc, bpp;
#pragma unroll
    for (int e = 0; e < 16; ++e) {
      int j = k0 + (e >> 3) * 16 + h * 8 + (e & 7);
      bc[e]  = lds_c[j * 16 + n];
      bpp[e] = lds_pp[j * 16 + n];
    }
#pragma unroll
    for (int t = 0; t < 4; ++t) {
      int m0 = (wave + 8 * t) * 16;
      // A fragment: element e -> K = k0 + (e>>3)*16 + h*8 + (e&7), row = n
      const float* kr = kernels + ((size_t)(p * SN + m0 + n)) * SN + k0 + h * 8;
      float4 s0 = *(const float4*)(kr);
      float4 s1 = *(const float4*)(kr + 4);
      float4 s2 = *(const float4*)(kr + 16);
      float4 s3 = *(const float4*)(kr + 20);
      float xv[16] = {s0.x,s0.y,s0.z,s0.w, s1.x,s1.y,s1.z,s1.w,
                      s2.x,s2.y,s2.z,s2.w, s3.x,s3.y,s3.z,s3.w};
      v16h aadj, asmx;
#pragma unroll
      for (int e = 0; e < 16; ++e) {
        float x = xv[e];
        aadj[e] = (_Float16)(x > 0.f ? 1.f : 0.f);
        asmx[e] = (_Float16)(__expf(x - rm[t]) * rz[t]);
      }
      accA[t] = __builtin_amdgcn_wmma_f32_16x16x32_f16(
          false, aadj, false, bc,  (short)0, accA[t], false, false);
      accP[t] = __builtin_amdgcn_wmma_f32_16x16x32_f16(
          false, asmx, false, bpp, (short)0, accP[t], false, false);
    }
  }

  // accumulate scalar contributions. C/D layout: elem r -> (M = m0+r+8h, N = n)
  float trdeg = 0.f, tradj = 0.f, f2a = 0.f;
#pragma unroll
  for (int t = 0; t < 4; ++t) {
    int m0 = (wave + 8 * t) * 16;
#pragma unroll
    for (int r = 0; r < 8; ++r) {
      int m = m0 + r + 8 * h;
      float aval = accA[t][r];
      float pval = accP[t][r];
      if (n == DN) trdeg += aval;                     // column 10 = A@q
      if (n < DN) {
        float dpv = ws[OFF_DP + (size_t)(p * SN + m) * DN + n];
        tradj += dpv * aval;
        f2a   += dpv * __logf(pval + 1e-20f);
      }
    }
  }
#pragma unroll
  for (int off = 16; off; off >>= 1) {
    trdeg += __shfl_xor(trdeg, off, 32);
    tradj += __shfl_xor(tradj, off, 32);
    f2a   += __shfl_xor(f2a,   off, 32);
  }
  if (lane == 0) { red[wave][0] = trdeg; red[wave][1] = tradj; red[wave][2] = f2a; }
  __syncthreads();
  if (tid == 0) {
    float a = 0.f, b = 0.f, c = 0.f;
    for (int w = 0; w < 8; ++w) { a += red[w][0]; b += red[w][1]; c += red[w][2]; }
    ws[OFF_GRAPH + 3 * p + 0] = a;
    ws[OFF_GRAPH + 3 * p + 1] = b;
    ws[OFF_GRAPH + 3 * p + 2] = c;
  }
}

// -------------------------------------------------------------------------
// 5) deterministic final combine
// -------------------------------------------------------------------------
__global__ void final_kernel(const float* __restrict__ ws, float* __restrict__ out) {
  float ent = 0.f, g = 0.f, trdeg = 0.f, tradj = 0.f, f2s = 0.f;
  for (int i = 0; i < 32; ++i)   ent += ws[OFF_ENT + i];
  for (int i = 0; i < 1024; ++i) g   += ws[OFF_G1 + i];
  for (int i = 0; i < 16; ++i) {
    trdeg += ws[OFF_GRAPH + 3 * i + 0];
    tradj += ws[OFF_GRAPH + 3 * i + 1];
    f2s   += ws[OFF_GRAPH + 3 * i + 2];
  }
  const float inv = 1.0f / (float)DOCN;
  float entropy = ent * inv;
  float g1 = -g * inv;                 // g1 = mean(-sum_d dp*sq)
  float f3 = (trdeg - tradj) * inv;
  float f2 = f2s * inv;
  out[0] = 0.2f * f3 - 2000.0f * g1 - 0.2f * (f2 + entropy);
}

// -------------------------------------------------------------------------
extern "C" void kernel_launch(void* const* d_in, const int* in_sizes, int n_in,
                              void* d_out, int out_size, void* d_ws, size_t ws_size,
                              hipStream_t stream) {
  const float* exp_log_dtot = (const float*)d_in[0];
  const float* prior_pi     = (const float*)d_in[1];
  const float* kernels      = (const float*)d_in[2];
  const float* sp_count     = (const float*)d_in[3];
  const float* post_topic   = (const float*)d_in[4];
  const float* post_domain  = (const float*)d_in[5];
  // d_in[6] = partition: equal contiguous partitions are baked into indexing.
  float* ws  = (float*)d_ws;
  float* out = (float*)d_out;

  domain_kernel  <<<DOCN / 256,    256, 0, stream>>>(post_domain, ws);
  rowstats_kernel<<<(PN * SN) / 8, 256, 0, stream>>>(kernels, ws);
  decon_kernel   <<<DOCN / 8,      256, 0, stream>>>(post_topic, sp_count, exp_log_dtot, ws);
  graph_kernel   <<<PN,            256, 0, stream>>>(kernels, prior_pi, ws);
  final_kernel   <<<1,             1,   0, stream>>>(ws, out);
}
